// NavGCM_22110491640465
// MI455X (gfx1250) — matrix operand order repaired and checked by reading, not compiled, not run on hardware
//
#include <hip/hip_runtime.h>

// Problem constants (from setup_inputs)
#define BB   64
#define TAU  16
#define FF   256
#define LL   1024
#define FOUT 256
#define KNN  10

typedef float    v2f  __attribute__((ext_vector_type(2)));
typedef float    v8f  __attribute__((ext_vector_type(8)));
typedef _Float16 v16h __attribute__((ext_vector_type(16)));

// d_out layout (floats), tuple flattened: padded_output, new_x, new_pos, new_rot, lens
#define OFF_POUT   0
#define OFF_NEWX   (BB*TAU*FOUT)                  // 262144
#define OFF_NEWPOS (OFF_NEWX + BB*LL*FF)          // 17039360
#define OFF_NEWROT (OFF_NEWPOS + BB*LL*3)         // 17235968
#define OFF_LENS   (OFF_NEWROT + BB*LL)           // 17301504

// ---------------------------------------------------------------------------
// Kernel 1: new_x = old_x with window [T[b], T[b]+taus[b]) overwritten from x.
// Pure HBM streaming copy (134 MB) -> float4 vectorized.
// ---------------------------------------------------------------------------
__global__ void k_update_x(const float4* __restrict__ x,
                           const float4* __restrict__ old_x,
                           const int* __restrict__ taus, const int* __restrict__ T,
                           float4* __restrict__ new_x) {
    const int QW = FF / 4;
    int v = blockIdx.x * blockDim.x + threadIdx.x;   // [0, BB*LL*QW)
    int b  = v / (LL * QW);
    int r  = v - b * (LL * QW);
    int s  = r / QW;
    int fq = r - s * QW;
    int rel = s - T[b];
    float4 val;
    if (rel >= 0 && rel < taus[b]) val = x[(b * TAU + rel) * QW + fq];
    else                           val = old_x[v];
    new_x[v] = val;
}

// ---------------------------------------------------------------------------
// Kernel 2: new_pos/new_rot (bug-faithful: window copies from old_pos/old_rot
// at offset rel, not from incoming pos/rot) + lens = T + taus (as float).
// ---------------------------------------------------------------------------
__global__ void k_update_prl(const float* __restrict__ old_pos,
                             const float* __restrict__ old_rot,
                             const int* __restrict__ taus, const int* __restrict__ T,
                             float* __restrict__ new_pos, float* __restrict__ new_rot,
                             float* __restrict__ lens_out) {
    int v = blockIdx.x * blockDim.x + threadIdx.x;   // [0, BB*LL)
    int b = v / LL;
    int s = v - b * LL;
    int rel = s - T[b];
    bool w = (rel >= 0 && rel < taus[b]);
    int src = w ? (b * LL + rel) : v;
    new_pos[v * 3 + 0] = old_pos[src * 3 + 0];
    new_pos[v * 3 + 1] = old_pos[src * 3 + 1];
    new_pos[v * 3 + 2] = old_pos[src * 3 + 2];
    new_rot[v] = old_rot[src];
    if (s == 0) lens_out[b] = (float)(T[b] + taus[b]);
}

// ---------------------------------------------------------------------------
// Kernel 3: per batch b (one block, 16 waves x 32 lanes):
//   wave w: KNN(k=10) for query row t=w (node T[b]+w) over valid nodes,
//   then cooperative staging of A_self/A_agg [16 x 256] into LDS,
//   then wave w computes N-tile n0=16w of D = A_self@W_self + A_agg@W_nbr + b
//   via V_WMMA_F32_16X16X4_F32, relu, write padded_output (masked t>=taus).
// ---------------------------------------------------------------------------
__global__ __launch_bounds__(512)
void k_gcn(const float* __restrict__ newx, const float* __restrict__ newpos,
           const int* __restrict__ taus, const int* __restrict__ T,
           const float* __restrict__ Wself, const float* __restrict__ Wnbr,
           const float* __restrict__ bout, float* __restrict__ pout) {
    __shared__ float sSelf[16][FF + 4];   // +4 pad: spread stride-256 column
    __shared__ float sAgg[16][FF + 4];    //         reads across LDS banks
    __shared__ int   sNbr[16][KNN];

    const int b    = blockIdx.x;
    const int tid  = threadIdx.x;
    const int wave = tid >> 5;
    const int lane = tid & 31;
    const int Tb   = T[b];
    const int taub = taus[b];
    const int lenb = Tb + taub;

    // ---- KNN: wave handles query t = wave ------------------------------
    {
        const int t  = wave;
        const int gq = Tb + t;                         // always < LL
        const float* qp = newpos + ((size_t)b * LL + gq) * 3;
        const float qx = qp[0], qy = qp[1], qz = qp[2];

        float dist[KNN]; int idxr[KNN];
        #pragma unroll
        for (int q = 0; q < KNN; ++q) { dist[q] = 3.0e38f; idxr[q] = 0x7fffffff; }

        for (int j = lane; j < lenb; j += 32) {        // valid cols only
            if (j == gq) continue;                     // no self loop
            const float* pj = newpos + ((size_t)b * LL + j) * 3;
            float dx = pj[0] - qx, dy = pj[1] - qy, dz = pj[2] - qz;
            float d = dx * dx + dy * dy + dz * dz;
            if (d < dist[KNN - 1]) {                   // register insertion sort
                dist[KNN - 1] = d; idxr[KNN - 1] = j;
                #pragma unroll
                for (int q = KNN - 1; q >= 1; --q) {
                    if (dist[q] < dist[q - 1]) {
                        float td = dist[q]; dist[q] = dist[q - 1]; dist[q - 1] = td;
                        int   ti = idxr[q]; idxr[q] = idxr[q - 1]; idxr[q - 1] = ti;
                    }
                }
            }
        }
        // wave32 butterfly merge of 32 sorted-10 lists; pop global min 10x
        #pragma unroll 1
        for (int k = 0; k < KNN; ++k) {
            float bv = dist[0]; int bi = idxr[0]; int bl = lane;
            #pragma unroll
            for (int off = 16; off >= 1; off >>= 1) {
                float ov = __shfl_xor(bv, off, 32);
                int   oi = __shfl_xor(bi, off, 32);
                int   ol = __shfl_xor(bl, off, 32);
                if (ov < bv || (ov == bv && oi < bi)) { bv = ov; bi = oi; bl = ol; }
            }
            if (lane == 0) sNbr[t][k] = bi;
            if (lane == bl) {                          // winner pops its head
                #pragma unroll
                for (int q = 0; q < KNN - 1; ++q) { dist[q] = dist[q + 1]; idxr[q] = idxr[q + 1]; }
                dist[KNN - 1] = 3.0e38f; idxr[KNN - 1] = 0x7fffffff;
            }
        }
    }
    __syncthreads();

    // ---- Stage A_self / A_agg (mean of 10 neighbor rows) ----------------
    for (int idx = tid; idx < 16 * FF; idx += 512) {
        int t = idx >> 8, f = idx & (FF - 1);
        int gq = Tb + t;
        sSelf[t][f] = newx[((size_t)b * LL + gq) * FF + f];
        float acc = 0.f;
        #pragma unroll
        for (int k = 0; k < KNN; ++k)
            acc += newx[((size_t)b * LL + sNbr[t][k]) * FF + f];
        sAgg[t][f] = acc * (1.0f / KNN);
    }
    __syncthreads();

    // ---- WMMA GEMM: wave owns 16-col N tile -----------------------------
    const int n0  = wave * 16;
    const int m   = lane & 15;            // row (A) / col (B,D) within tile
    const int kk  = (lane >> 4) * 2;      // A/B K sub-offset per half-wave
    const int col = n0 + m;
    v8f acc = {};

#if defined(__gfx1250__) && __has_builtin(__builtin_amdgcn_wmma_f32_16x16x4_f32)
    for (int k0 = 0; k0 < FF; k0 += 4) {
        v2f aS, aG, bS, bG;
        // A 16x4 f32 layout: lanes0-15 VGPR{0,1}=K{0,1}; lanes16-31 K{2,3}
        aS.x = sSelf[m][k0 + kk];  aS.y = sSelf[m][k0 + kk + 1];
        aG.x = sAgg[m][k0 + kk];   aG.y = sAgg[m][k0 + kk + 1];
        // B 4x16 f32: row K striped across lanes within a VGPR (mirror of A)
        bS.x = Wself[(k0 + kk) * FOUT + col];  bS.y = Wself[(k0 + kk + 1) * FOUT + col];
        bG.x = Wnbr [(k0 + kk) * FOUT + col];  bG.y = Wnbr [(k0 + kk + 1) * FOUT + col];
        acc = __builtin_amdgcn_wmma_f32_16x16x4_f32(false, aS, false, bS, (short)0, acc, false, false);
        acc = __builtin_amdgcn_wmma_f32_16x16x4_f32(false, aG, false, bG, (short)0, acc, false, false);
    }
#elif defined(__gfx1250__)
    // Fallback: codegen-confirmed f16 WMMA (K=32), convert f32->f16 on load.
    for (int k0 = 0; k0 < FF; k0 += 32) {
        v16h aS, aG, bS, bG;
        #pragma unroll
        for (int h = 0; h < 16; ++h) {
            int v = h >> 1, s = h & 1;
            // A 16x32 f16: lanes0-15 K=0..15, lanes16-31 K=16..31
            int kA = v * 2 + s + (lane >> 4) * 16;
            aS[h] = (_Float16)sSelf[m][k0 + kA];
            aG[h] = (_Float16)sAgg[m][k0 + kA];
            // B 32x16 f16: lanes0-15 K=0..15, lanes16-31 K=16..31
            int kB = v * 2 + s + (lane >> 4) * 16;
            bS[h] = (_Float16)Wself[(k0 + kB) * FOUT + col];
            bG[h] = (_Float16)Wnbr [(k0 + kB) * FOUT + col];
        }
        acc = __builtin_amdgcn_wmma_f32_16x16x32_f16(false, aS, false, bS, (short)0, acc, false, false);
        acc = __builtin_amdgcn_wmma_f32_16x16x32_f16(false, aG, false, bG, (short)0, acc, false, false);
    }
#else
    // Host-pass / non-gfx1250 parse path (never executed on device of interest)
    #pragma unroll 1
    for (int r = 0; r < 8; ++r) {
        int row = r + ((lane >> 4) << 3);
        float s = 0.f;
        for (int f = 0; f < FF; ++f)
            s += sSelf[row][f] * Wself[f * FOUT + col] + sAgg[row][f] * Wnbr[f * FOUT + col];
        acc[r] = s;
    }
#endif

    // ---- epilogue: bias + relu + mask t>=taus, write padded_output ------
    const float bias = bout[col];
    #pragma unroll
    for (int r = 0; r < 8; ++r) {
        int row = r + ((lane >> 4) << 3);     // C/D: VGPR r -> M = r + 8*(lane>=16)
        float v = acc[r] + bias;
        v = v > 0.f ? v : 0.f;
        if (row >= taub) v = 0.f;
        pout[((size_t)b * TAU + row) * FOUT + col] = v;
    }
}

// ---------------------------------------------------------------------------
extern "C" void kernel_launch(void* const* d_in, const int* in_sizes, int n_in,
                              void* d_out, int out_size, void* d_ws, size_t ws_size,
                              hipStream_t stream) {
    (void)in_sizes; (void)n_in; (void)out_size; (void)d_ws; (void)ws_size;
    const float* x       = (const float*)d_in[0];
    // d_in[1] = pos, d_in[2] = rot: intentionally unused (bug-faithful ref)
    const float* old_x   = (const float*)d_in[3];
    const float* old_pos = (const float*)d_in[4];
    const float* old_rot = (const float*)d_in[5];
    const int*   taus    = (const int*)d_in[6];
    const int*   T       = (const int*)d_in[7];
    const float* Wself   = (const float*)d_in[8];
    const float* Wnbr    = (const float*)d_in[9];
    const float* bout    = (const float*)d_in[10];

    float* out    = (float*)d_out;
    float* pout   = out + OFF_POUT;
    float* newx   = out + OFF_NEWX;
    float* newpos = out + OFF_NEWPOS;
    float* newrot = out + OFF_NEWROT;
    float* lens   = out + OFF_LENS;

    k_update_x<<<(BB * LL * (FF / 4)) / 256, 256, 0, stream>>>(
        (const float4*)x, (const float4*)old_x, taus, T, (float4*)newx);
    k_update_prl<<<(BB * LL) / 256, 256, 0, stream>>>(
        old_pos, old_rot, taus, T, newpos, newrot, lens);
    k_gcn<<<BB, 512, 0, stream>>>(newx, newpos, taus, T, Wself, Wnbr, bout, pout);
}
